// LIAFRCell_8332236554268
// MI455X (gfx1250) — matrix operand order repaired and checked by reading, not compiled, not run on hardware
//
#include <hip/hip_runtime.h>

typedef __bf16 bf16;
typedef __attribute__((ext_vector_type(16))) __bf16 v16bf;
typedef __attribute__((ext_vector_type(8)))  __bf16 v8bf;
typedef __attribute__((ext_vector_type(8)))  float  v8f;

#define B_  256
#define T_  32
#define IN_ 1024
#define H_  1024
#define THRESH 0.25f
#define SCAN_NB 128   // persistent scan: 4 x 32 blocks, must stay co-resident

#if __has_builtin(__builtin_amdgcn_global_load_async_to_lds_b128)
#define HAVE_ASYNC_LDS 1
// Builtin signature (from hipcc diagnostic): params are pointers to
// gcc-vector int4; arg0 in AS1 (__device__), arg1 in AS3 (__shared__).
typedef int v4i_t __attribute__((vector_size(16)));
typedef __attribute__((address_space(1))) v4i_t g_v4i;
typedef __attribute__((address_space(3))) v4i_t l_v4i;
#endif

// ---- fragment loaders (contiguous K order, shared by A and B) --------------------

// A-matrix 16x32 bf16 fragment from a row-major fp32 row (K-contiguous).
// lane (0-15): K = k0 + {0..15};  lane (16-31): K = k0 + {16..31}
__device__ inline v16bf load_a_frag_f32(const float* __restrict__ row, int k0, int khalf) {
  const float* p = row + k0 + khalf * 16;
  v8f lo = *(const v8f*)(p);
  v8f hi = *(const v8f*)(p + 8);
  v16bf r;
#pragma unroll
  for (int e = 0; e < 8; ++e) { r[e] = (bf16)lo[e]; r[e + 8] = (bf16)hi[e]; }
  return r;
}

// Same A fragment but from a bf16 source (recurrent state path): two 16B loads.
__device__ inline v16bf load_a_frag_bf16(const bf16* __restrict__ row, int k0, int khalf) {
  const bf16* p = row + k0 + khalf * 16;
  v8bf lo = *(const v8bf*)(p);
  v8bf hi = *(const v8bf*)(p + 8);
  v16bf r;
#pragma unroll
  for (int e = 0; e < 8; ++e) { r[e] = lo[e]; r[e + 8] = hi[e]; }
  return r;
}

// B-matrix 32x16 bf16 fragment from global: one 32B load per lane.
__device__ inline v16bf load_b_frag_bf16(const bf16* __restrict__ row, int k0, int khalf) {
  return *(const v16bf*)(row + k0 + khalf * 16);
}

// ---- device-wide barrier for the persistent scan kernel -----------------
__device__ inline void grid_sync(unsigned* __restrict__ cnt, unsigned target) {
  __threadfence();          // release this block's v-state writes
  __syncthreads();
  if (threadIdx.x == 0) {
    __hip_atomic_fetch_add(cnt, 1u, __ATOMIC_RELEASE, __HIP_MEMORY_SCOPE_AGENT);
    while (__hip_atomic_load(cnt, __ATOMIC_ACQUIRE, __HIP_MEMORY_SCOPE_AGENT) < target) {
      __builtin_amdgcn_s_sleep(1);
    }
  }
  __syncthreads();
  __threadfence();          // acquire other blocks' v-state writes
}

// ---- prep: weights fp32->bf16, zero bf16 v-state, reset barrier ---------
__global__ __launch_bounds__(256)
void prep_kernel(const float* __restrict__ Win, const float* __restrict__ Wv,
                 bf16* __restrict__ WinB, bf16* __restrict__ WvB,
                 bf16* __restrict__ V0, bf16* __restrict__ V1,
                 unsigned* __restrict__ cnt) {
  int i = blockIdx.x * blockDim.x + threadIdx.x;
  if (i < H_ * IN_) {
    WinB[i] = (bf16)Win[i];
    WvB[i]  = (bf16)Wv[i];
  }
  if (i < B_ * H_) { V0[i] = (bf16)0.0f; V1[i] = (bf16)0.0f; }
  if (i == 0) *cnt = 0u;
}

// ---- pre-projection GEMM: Out[8192,1024] = X * WinB^T + b ---------------
// block tile 128x128, 8 waves as 4(M) x 2(N), wave tile 32x64 (2x4 frags)
__global__ __launch_bounds__(256)
void gemm_pre_kernel(const float* __restrict__ X, const bf16* __restrict__ Wb,
                     const float* __restrict__ bias, float* __restrict__ Out) {
  const int lane  = threadIdx.x & 31;
  const int wave  = threadIdx.x >> 5;
  const int wm    = wave >> 1;
  const int wn    = wave & 1;
  const int mbase = blockIdx.x * 128 + wm * 32;
  const int nbase = blockIdx.y * 128 + wn * 64;
  const int lm    = lane & 15;
  const int khalf = lane >> 4;

  v8f acc[2][4];
#pragma unroll
  for (int i = 0; i < 2; ++i)
#pragma unroll
    for (int j = 0; j < 4; ++j) acc[i][j] = (v8f)0.0f;

  const float* arow[2];
#pragma unroll
  for (int i = 0; i < 2; ++i) arow[i] = X + (size_t)(mbase + i * 16 + lm) * IN_;
  const bf16* brow[4];
#pragma unroll
  for (int j = 0; j < 4; ++j) brow[j] = Wb + (size_t)(nbase + j * 16 + lm) * IN_;

  for (int k0 = 0; k0 < IN_; k0 += 32) {
    v16bf a[2], b[4];
#pragma unroll
    for (int i = 0; i < 2; ++i) a[i] = load_a_frag_f32(arow[i], k0, khalf);
#pragma unroll
    for (int j = 0; j < 4; ++j) b[j] = load_b_frag_bf16(brow[j], k0, khalf);
#pragma unroll
    for (int i = 0; i < 2; ++i)
#pragma unroll
      for (int j = 0; j < 4; ++j)
        acc[i][j] = __builtin_amdgcn_wmma_f32_16x16x32_bf16(
            false, a[i], false, b[j], (short)0, acc[i][j], false, false);
  }

#pragma unroll
  for (int i = 0; i < 2; ++i)
#pragma unroll
    for (int j = 0; j < 4; ++j) {
      const int n = nbase + j * 16 + lm;
      const float bn = bias[n];
#pragma unroll
      for (int r = 0; r < 8; ++r) {
        const int m = mbase + i * 16 + khalf * 8 + r;
        Out[(size_t)m * H_ + n] = acc[i][j][r] + bn;   // pre, staged in d_out
      }
    }
}

// ---- persistent recurrent scan: all 32 timesteps in ONE kernel ----------
// grid 4(M) x 32(N) = 128 blocks; block tile 64x32; 8 waves as 4(M) x 2(N);
// wave tile 16x16. W_v panel (32 rows x 1024 K = 64 KB) staged in LDS ONCE
// (async global->LDS where available), pre-swizzled into WMMA B-fragment
// order, reused by all 32 timesteps.
__global__ __launch_bounds__(256)
void lif_scan_kernel(const bf16* __restrict__ Wv, const float* __restrict__ bias,
                     const float* __restrict__ decayp,
                     float* __restrict__ InOut,      // [B,T,H]: reads pre_t, writes out_t
                     bf16* __restrict__ v0, bf16* __restrict__ v1,
                     unsigned* __restrict__ cnt) {
  __shared__ bf16 Bs[2 * 32 * 32 * 16];   // 64 KB: [ngroup][kchunk][lane][16 bf16]

  const int lane  = threadIdx.x & 31;
  const int wave  = threadIdx.x >> 5;
  const int wm    = wave >> 1;                   // 0..3
  const int wn    = wave & 1;                    // 0..1
  const int mbase = blockIdx.x * 64 + wm * 16;
  const int n0    = blockIdx.y * 32;
  const int lm    = lane & 15;
  const int khalf = lane >> 4;

  // one-time staging: 2048 fragment-lane chunks of 32B, 8 per thread
  for (int c = threadIdx.x; c < 2 * 32 * 32; c += 256) {
    const int cl  = c & 31;                      // lane within fragment
    const int kc  = (c >> 5) & 31;               // K chunk (32 K each)
    const int ng  = c >> 10;                     // N group of 16
    const int row = n0 + ng * 16 + (cl & 15);
    const int kk  = kc * 32 + (cl >> 4) * 16;
    const bf16* gsrc = &Wv[(size_t)row * H_ + kk];
    bf16*       ldst = &Bs[(size_t)c * 16];
#ifdef HAVE_ASYNC_LDS
    // INST_OFFSET applies to both global and LDS addresses (ISA §10 async):
    // two b128 issues cover the 32B chunk, tracked by ASYNCcnt.
    __builtin_amdgcn_global_load_async_to_lds_b128(
        (g_v4i*)gsrc, (l_v4i*)ldst, 0, 0);
    __builtin_amdgcn_global_load_async_to_lds_b128(
        (g_v4i*)gsrc, (l_v4i*)ldst, 16, 0);
#else
    *(v16bf*)ldst = *(const v16bf*)gsrc;
#endif
  }
#ifdef HAVE_ASYNC_LDS
#if __has_builtin(__builtin_amdgcn_s_wait_asynccnt)
  __builtin_amdgcn_s_wait_asynccnt(0);
#else
  asm volatile("s_wait_asynccnt 0x0" ::: "memory");
#endif
#endif
  __syncthreads();

  const float dec = decayp[0];
  const int   n   = n0 + wn * 16 + lm;
  const float bn  = bias[n];

  for (int t = 0; t < T_; ++t) {
    const bf16* Vp = (t & 1) ? v1 : v0;
    bf16*       Vn = (t & 1) ? v0 : v1;
    const bf16* arow = Vp + (size_t)(mbase + lm) * H_;

    v8f acc = (v8f)0.0f;
    for (int kc = 0; kc < 32; ++kc) {
      v16bf b = *(const v16bf*)&Bs[(((size_t)wn * 32 + kc) * 32 + lane) * 16];
      v16bf a = load_a_frag_bf16(arow, kc * 32, khalf);
      acc = __builtin_amdgcn_wmma_f32_16x16x32_bf16(
          false, a, false, b, (short)0, acc, false, false);
    }

#pragma unroll
    for (int r = 0; r < 8; ++r) {
      const int m = mbase + khalf * 8 + r;             // batch index
      const size_t oidx = ((size_t)m * T_ + t) * H_ + n;
      float v = acc[r] + InOut[oidx] + bn;             // pre_t read
      float fire = (v > THRESH) ? 1.0f : 0.0f;
      float o = 1.0507009873554805f *
                ((v > 0.0f) ? v : 1.6732632423543772f * (__expf(v) - 1.0f));
      InOut[oidx] = o;                                  // selu output
      Vn[(size_t)m * H_ + n] = (bf16)(dec * (1.0f - fire) * v);
    }

    grid_sync(cnt, (unsigned)(SCAN_NB * (t + 1)));      // step boundary
  }
}

extern "C" void kernel_launch(void* const* d_in, const int* in_sizes, int n_in,
                              void* d_out, int out_size, void* d_ws, size_t ws_size,
                              hipStream_t stream) {
  const float* x     = (const float*)d_in[0];
  const float* W_in  = (const float*)d_in[1];
  const float* b_in  = (const float*)d_in[2];
  const float* W_v   = (const float*)d_in[3];
  const float* b_v   = (const float*)d_in[4];
  const float* decay = (const float*)d_in[5];
  float* out = (float*)d_out;

  char* ws = (char*)d_ws;
  bf16*     WinB = (bf16*)ws;                                        // 2 MB
  bf16*     WvB  = (bf16*)(ws + (size_t)H_ * IN_ * 2);               // 2 MB
  bf16*     vb0  = (bf16*)(ws + (size_t)H_ * IN_ * 4);               // 512 KB
  bf16*     vb1  = vb0 + (size_t)B_ * H_;                            // 512 KB
  unsigned* cnt  = (unsigned*)(ws + (size_t)H_ * IN_ * 4 + (size_t)B_ * H_ * 4);

  // 1) weights -> bf16, zero v-state, reset device barrier (every call)
  prep_kernel<<<(H_ * IN_ + 255) / 256, 256, 0, stream>>>(W_in, W_v, WinB, WvB,
                                                          vb0, vb1, cnt);

  // 2) input projection for all timesteps, staged into d_out
  gemm_pre_kernel<<<dim3((B_ * T_) / 128, H_ / 128), 256, 0, stream>>>(x, WinB, b_in, out);

  // 3) one persistent kernel runs the entire 32-step scan (LDS-resident W_v)
  lif_scan_kernel<<<dim3(B_ / 64, H_ / 32), 256, 0, stream>>>(WvB, b_v, decay, out,
                                                              vb0, vb1, cnt);
}
